// Wav2Vec2MultiHeadAttention_37314675867712
// MI455X (gfx1250) — matrix operand order repaired and checked by reading, not compiled, not run on hardware
//
#include <hip/hip_runtime.h>
#include <hip/hip_bf16.h>

// ---------------------------------------------------------------------------
// Types for CDNA5 WMMA (wave32): v16bf A/B fragments, v8f C/D accumulators.
// ---------------------------------------------------------------------------
typedef __bf16 bf16;
typedef __attribute__((ext_vector_type(16))) __bf16 v16bf;
typedef __attribute__((ext_vector_type(8)))  __bf16 bf16x8;
typedef __attribute__((ext_vector_type(4)))  __bf16 bf16x4;
typedef __attribute__((ext_vector_type(8)))  float  v8f;

// ---------------------------------------------------------------------------
// Async global->LDS (gfx1250 GLOBAL_LOAD_ASYNC_TO_LDS_B128, ASYNCcnt) via
// inline asm (ISA 15.18.3 op 98; GV mode: vdst = LDS addr, vaddr = 64b addr,
// "off" = null SADDR). LDS operand = low 32 bits of the generic pointer
// (aperture rule, ISA 10.2). ASYNCcnt completes in order, so a deferred
// s_wait_asynccnt N implements double-buffered pipelining.
// ---------------------------------------------------------------------------
static __device__ __forceinline__ void async_to_lds_b128(const void* g, void* l) {
  asm volatile("global_load_async_to_lds_b128 %0, %1, off"
               :
               : "v"((unsigned int)(unsigned long long)l), "v"(g)
               : "memory");
}
static __device__ __forceinline__ void wait_async_le4() {
  asm volatile("s_wait_asynccnt 4" ::: "memory");
}
static __device__ __forceinline__ void wait_async_0() {
  asm volatile("s_wait_asynccnt 0" ::: "memory");
}

// Problem constants (match reference)
constexpr int E    = 1024;
constexpr int H    = 16;
constexpr int Dh   = 64;
constexpr int SEQ  = 2048;
constexpr int Bb   = 2;
constexpr int Mrows = Bb * SEQ;   // 4096

// Load a 16-element bf16 A/B fragment as two contiguous 16-byte loads.
static __device__ __forceinline__ v16bf ldfrag(const bf16* p0, const bf16* p1) {
  bf16x8 lo = *(const bf16x8*)p0;
  bf16x8 hi = *(const bf16x8*)p1;
  return __builtin_shufflevector(lo, hi, 0,1,2,3,4,5,6,7,8,9,10,11,12,13,14,15);
}

static __device__ __forceinline__ v8f wmma_bf16(v16bf a, v16bf b, v8f c) {
  return __builtin_amdgcn_wmma_f32_16x16x32_bf16(false, a, false, b,
                                                 (short)0, c, false, false);
}

// ---------------------------------------------------------------------------
// fp32 -> bf16 elementwise (vectorized x4)
// ---------------------------------------------------------------------------
__global__ void __launch_bounds__(256)
f32_to_bf16_vec(const float* __restrict__ in, bf16* __restrict__ out, int n) {
  int i = (blockIdx.x * 256 + threadIdx.x) * 4;
  if (i + 3 < n) {
    const float4 v = *(const float4*)(in + i);
    bf16x4 o4 = { (bf16)v.x, (bf16)v.y, (bf16)v.z, (bf16)v.w };
    *(bf16x4*)(out + i) = o4;
  }
}

// ---------------------------------------------------------------------------
// Tiled transpose: fp32 W[K][N] -> bf16 W^T[N][K] (coalesced both sides)
// ---------------------------------------------------------------------------
__global__ void __launch_bounds__(256)
transpose_f32_to_bf16(const float* __restrict__ in, bf16* __restrict__ out,
                      int K, int N) {
  __shared__ bf16 t[32][33];
  const int tx = threadIdx.x & 31;
  const int ty = threadIdx.x >> 5;      // 0..7
  const int k0 = blockIdx.y * 32;
  const int n0 = blockIdx.x * 32;
  #pragma unroll
  for (int i = 0; i < 32; i += 8)
    t[ty + i][tx] = (bf16)in[(size_t)(k0 + ty + i) * N + n0 + tx];
  __syncthreads();
  #pragma unroll
  for (int i = 0; i < 32; i += 8)
    out[(size_t)(n0 + ty + i) * K + k0 + tx] = t[tx][ty + i];
}

// ---------------------------------------------------------------------------
// GEMM: C[M,N] = A[M,K](bf16,row) * B^T[N,K](bf16,row) + bias
// Block = 256 threads (8 waves), tile 128x128x32, wave grid 2x4,
// each wave computes 64x32 via 4x2 WMMA 16x16x32 bf16 accum f32.
// Global->LDS staging: double-buffered async-LDS DMA; the s_wait_asynccnt 4
// drains only the previous tile's 4 ops, so this tile's DMA overlaps the
// previous tile's WMMAs (ASYNCcnt is in-order).
// mode: 0 = bf16 [M,N]; 1 = bf16 [B,H,S,D]; 2 = bf16 [B,H,D,S]; 3 = f32 [M,N]
// ---------------------------------------------------------------------------
__global__ void __launch_bounds__(256)
gemm_wmma_bf16(const bf16* __restrict__ A, const bf16* __restrict__ BT,
               const float* __restrict__ bias, void* __restrict__ outp,
               int M, int N, int K, int mode) {
  __shared__ bf16 lA[2][128][32];   // double-buffered A tile, row-major
  __shared__ bf16 lB[2][128][32];   // double-buffered B^T tile: [n][k]
  const int tid  = threadIdx.x;
  const int lane = tid & 31;
  const int w    = tid >> 5;
  const int wm   = w >> 2;                 // 0..1
  const int wn   = w & 3;                  // 0..3
  const int m0   = blockIdx.y * 128;
  const int n0   = blockIdx.x * 128;
  const int ml   = lane & 15;
  const int kbA  = (lane < 16) ? 0 : 8;    // A frag K-base per half-wave
  const int kbB  = (lane < 16) ? 0 : 16;   // unused; B frags use kbA as well

  v8f acc[4][2];
  #pragma unroll
  for (int i = 0; i < 4; i++)
    #pragma unroll
    for (int j = 0; j < 2; j++)
      #pragma unroll
      for (int r = 0; r < 8; r++) acc[i][j][r] = 0.0f;

  const int lrow  = tid >> 1;        // 0..127
  const int lhalf = (tid & 1) * 16;  // 0 or 16

  // issue this thread's 4 async B128 fills for K-tile k0 into buffer `buf`
  auto issue_tile = [&](int k0, int buf) {
    async_to_lds_b128(&A [(size_t)(m0 + lrow) * K + k0 + lhalf],     &lA[buf][lrow][lhalf]);
    async_to_lds_b128(&A [(size_t)(m0 + lrow) * K + k0 + lhalf + 8], &lA[buf][lrow][lhalf + 8]);
    async_to_lds_b128(&BT[(size_t)(n0 + lrow) * K + k0 + lhalf],     &lB[buf][lrow][lhalf]);
    async_to_lds_b128(&BT[(size_t)(n0 + lrow) * K + k0 + lhalf + 8], &lB[buf][lrow][lhalf + 8]);
  };

  const int nT = K / 32;
  issue_tile(0, 0);                 // prologue: fill buffer 0

  for (int t = 0; t < nT; ++t) {
    const int buf = t & 1;
    if (t + 1 < nT) {
      issue_tile((t + 1) * 32, buf ^ 1);  // prefetch next tile (stays in flight)
      wait_async_le4();                   // drain only the current tile's 4 ops
    } else {
      wait_async_0();
    }
    __syncthreads();                      // current buffer filled by all waves

    v16bf bfrag[2];
    #pragma unroll
    for (int j = 0; j < 2; j++) {
      const bf16* bp = &lB[buf][wn * 32 + j * 16 + ml][0];
      bfrag[j] = ldfrag(bp + kbA, bp + 16 + kbA);
    }
    #pragma unroll
    for (int i = 0; i < 4; i++) {
      const bf16* ap = &lA[buf][wm * 64 + i * 16 + ml][0];
      v16bf afrag = ldfrag(ap + kbA, ap + 16 + kbA);
      #pragma unroll
      for (int j = 0; j < 2; j++)
        acc[i][j] = wmma_bf16(afrag, bfrag[j], acc[i][j]);
    }
    __syncthreads();                      // all waves done reading `buf` before
                                          // it becomes the DMA target at t+1
  }

  // Epilogue: C/D layout -> lane (l&15)=col, row = r + 8*(l>=16)
  const int rsel8 = (lane >> 4) * 8;
  #pragma unroll
  for (int i = 0; i < 4; i++) {
    #pragma unroll
    for (int j = 0; j < 2; j++) {
      const int col = n0 + wn * 32 + j * 16 + ml;
      const float bv = bias[col];
      #pragma unroll
      for (int r = 0; r < 8; r++) {
        const int row = m0 + wm * 64 + i * 16 + r + rsel8;
        const float v = acc[i][j][r] + bv;
        if (mode == 0) {
          ((bf16*)outp)[(size_t)row * N + col] = (bf16)v;
        } else if (mode == 1) {          // [B,H,S,D]
          const int bb = row >> 11, s = row & (SEQ - 1);
          const int hh = col >> 6,  d = col & (Dh - 1);
          ((bf16*)outp)[(((size_t)(bb * H + hh)) * SEQ + s) * Dh + d] = (bf16)v;
        } else if (mode == 2) {          // [B,H,D,S]  (V transposed)
          const int bb = row >> 11, s = row & (SEQ - 1);
          const int hh = col >> 6,  d = col & (Dh - 1);
          ((bf16*)outp)[(((size_t)(bb * H + hh)) * Dh + d) * SEQ + s] = (bf16)v;
        } else {                         // fp32 output
          ((float*)outp)[(size_t)row * N + col] = v;
        }
      }
    }
  }
}

// ---------------------------------------------------------------------------
// Flash attention: grid = B*H*(SEQ/64), block = 128 (4 waves).
// Each wave owns 16 query rows; streams KV in 64-key blocks with online
// softmax. QK^T and P*V via WMMA bf16; P converted C-layout -> A-layout
// through per-wave LDS (DS ops are in-order within a wave).
// ---------------------------------------------------------------------------
__global__ void __launch_bounds__(128)
flash_attn_wmma(const bf16* __restrict__ Q, const bf16* __restrict__ Kb,
                const bf16* __restrict__ Vt, const float* __restrict__ mask,
                bf16* __restrict__ ctx) {
  __shared__ bf16 Pl[4][16][64];             // per-wave P tile (bf16)
  const int qblk = blockIdx.x & ((SEQ / 64) - 1);
  const int bh   = blockIdx.x >> 5;          // SEQ/64 == 32
  const int b    = bh >> 4;                  // /H
  const int h    = bh & (H - 1);
  const int lane = threadIdx.x & 31;
  const int w    = threadIdx.x >> 5;
  const int q0   = qblk * 64 + w * 16;
  const int ml   = lane & 15;
  const int kbA  = (lane < 16) ? 0 : 8;
  const int kbB  = (lane < 16) ? 0 : 16;
  const int rsel8 = (lane >> 4) * 8;

  const bf16* Qbase = Q  + (size_t)bh * SEQ * Dh;
  const bf16* Kbase = Kb + (size_t)bh * SEQ * Dh;
  const bf16* Vbase = Vt + (size_t)bh * Dh * SEQ;

  // Q A-fragments (16x64 = two 16x32 frags), loaded once, AMD table k order per lane
  const bf16* qrow = Qbase + (size_t)(q0 + ml) * Dh;
  const v16bf qf0 = ldfrag(qrow + kbA,      qrow + 16 + kbA);
  const v16bf qf1 = ldfrag(qrow + 32 + kbA, qrow + 48 + kbA);

  v8f o[4];
  float mst[8], lst[8];
  #pragma unroll
  for (int j = 0; j < 4; j++)
    #pragma unroll
    for (int r = 0; r < 8; r++) o[j][r] = 0.0f;
  #pragma unroll
  for (int r = 0; r < 8; r++) { mst[r] = -3.0e38f; lst[r] = 0.0f; }

  for (int kv0 = 0; kv0 < SEQ; kv0 += 64) {
    // ---- S = Q * K^T (scaled) ----
    v8f sa[4];
    #pragma unroll
    for (int j = 0; j < 4; j++)
      #pragma unroll
      for (int r = 0; r < 8; r++) sa[j][r] = 0.0f;

    #pragma unroll
    for (int j = 0; j < 4; j++) {
      const bf16* krow = Kbase + (size_t)(kv0 + j * 16 + ml) * Dh;
      v16bf kf0 = ldfrag(krow + kbA,      krow + 16 + kbA);
      v16bf kf1 = ldfrag(krow + 32 + kbA, krow + 48 + kbA);
      sa[j] = wmma_bf16(qf0, kf0, sa[j]);
      sa[j] = wmma_bf16(qf1, kf1, sa[j]);
    }

    // scale + additive mask (per key column)
    float mk[4];
    #pragma unroll
    for (int j = 0; j < 4; j++)
      mk[j] = (1.0f - mask[(size_t)b * SEQ + kv0 + j * 16 + ml]) * -10000.0f;
    #pragma unroll
    for (int j = 0; j < 4; j++)
      #pragma unroll
      for (int r = 0; r < 8; r++) sa[j][r] = sa[j][r] * 0.125f + mk[j];

    // ---- online softmax (rows live in 16-lane halves) ----
    float rmax[8], rsum[8], corr[8];
    #pragma unroll
    for (int r = 0; r < 8; r++)
      rmax[r] = fmaxf(fmaxf(sa[0][r], sa[1][r]), fmaxf(sa[2][r], sa[3][r]));
    #pragma unroll
    for (int off = 1; off < 16; off <<= 1)
      #pragma unroll
      for (int r = 0; r < 8; r++)
        rmax[r] = fmaxf(rmax[r], __shfl_xor(rmax[r], off, 32));
    #pragma unroll
    for (int r = 0; r < 8; r++) {
      const float mn = fmaxf(mst[r], rmax[r]);
      corr[r] = __expf(mst[r] - mn);
      mst[r] = mn;
      rsum[r] = 0.0f;
    }
    #pragma unroll
    for (int j = 0; j < 4; j++)
      #pragma unroll
      for (int r = 0; r < 8; r++) {
        const float p = __expf(sa[j][r] - mst[r]);
        sa[j][r] = p;
        rsum[r] += p;
      }
    #pragma unroll
    for (int off = 1; off < 16; off <<= 1)
      #pragma unroll
      for (int r = 0; r < 8; r++)
        rsum[r] += __shfl_xor(rsum[r], off, 32);
    #pragma unroll
    for (int r = 0; r < 8; r++) lst[r] = lst[r] * corr[r] + rsum[r];
    #pragma unroll
    for (int j = 0; j < 4; j++)
      #pragma unroll
      for (int r = 0; r < 8; r++) o[j][r] *= corr[r];

    // ---- P: C-layout f32 -> A-layout bf16 via per-wave LDS ----
    #pragma unroll
    for (int j = 0; j < 4; j++)
      #pragma unroll
      for (int r = 0; r < 8; r++)
        Pl[w][r + rsel8][j * 16 + ml] = (bf16)sa[j][r];
    const bf16* prow = &Pl[w][ml][0];
    v16bf pf0 = ldfrag(prow + kbA,      prow + 16 + kbA);
    v16bf pf1 = ldfrag(prow + 32 + kbA, prow + 48 + kbA);

    // ---- O += P * V (V stored [B,H,D,S]: contiguous along keys) ----
    #pragma unroll
    for (int j = 0; j < 4; j++) {
      const bf16* vrow = Vbase + (size_t)(j * 16 + ml) * SEQ + kv0;
      v16bf vf0 = ldfrag(vrow + kbA,      vrow + 16 + kbA);
      v16bf vf1 = ldfrag(vrow + 32 + kbA, vrow + 48 + kbA);
      o[j] = wmma_bf16(pf0, vf0, o[j]);
      o[j] = wmma_bf16(pf1, vf1, o[j]);
    }
  }

  // normalize and store ctx as bf16 [B,S,E]
  float rinv[8];
  #pragma unroll
  for (int r = 0; r < 8; r++) rinv[r] = 1.0f / lst[r];
  #pragma unroll
  for (int j = 0; j < 4; j++)
    #pragma unroll
    for (int r = 0; r < 8; r++) {
      const int srow = q0 + r + rsel8;
      const int col  = h * Dh + j * 16 + ml;
      ctx[((size_t)(b * SEQ + srow)) * E + col] = (bf16)(o[j][r] * rinv[r]);
    }
}

// ---------------------------------------------------------------------------
// Launch
// ---------------------------------------------------------------------------
extern "C" void kernel_launch(void* const* d_in, const int* in_sizes, int n_in,
                              void* d_out, int out_size, void* d_ws, size_t ws_size,
                              hipStream_t stream) {
  const float* X    = (const float*)d_in[0];
  const float* mask = (const float*)d_in[1];
  const float* Wq   = (const float*)d_in[2];
  const float* bq   = (const float*)d_in[3];
  const float* Wk   = (const float*)d_in[4];
  const float* bk   = (const float*)d_in[5];
  const float* Wv   = (const float*)d_in[6];
  const float* bv   = (const float*)d_in[7];
  const float* Wo   = (const float*)d_in[8];
  const float* bo   = (const float*)d_in[9];

  // workspace layout (bf16), total ~48 MB
  bf16* Xbf = (bf16*)d_ws;
  bf16* WqT = Xbf + (size_t)Mrows * E;
  bf16* WkT = WqT + (size_t)E * E;
  bf16* WvT = WkT + (size_t)E * E;
  bf16* WoT = WvT + (size_t)E * E;
  bf16* Qb  = WoT + (size_t)E * E;                    // [B,H,S,D]
  bf16* Kbf = Qb  + (size_t)Bb * H * SEQ * Dh;        // [B,H,S,D]
  bf16* Vtb = Kbf + (size_t)Bb * H * SEQ * Dh;        // [B,H,D,S]
  bf16* Ctx = Vtb + (size_t)Bb * H * SEQ * Dh;        // [B*S, E]

  // 1) convert activations to bf16
  f32_to_bf16_vec<<<(Mrows * E) / 1024, 256, 0, stream>>>(X, Xbf, Mrows * E);

  // 2) transpose+convert the four weight matrices: W[K][N] -> W^T[N][K] bf16
  dim3 tg(E / 32, E / 32);
  transpose_f32_to_bf16<<<tg, 256, 0, stream>>>(Wq, WqT, E, E);
  transpose_f32_to_bf16<<<tg, 256, 0, stream>>>(Wk, WkT, E, E);
  transpose_f32_to_bf16<<<tg, 256, 0, stream>>>(Wv, WvT, E, E);
  transpose_f32_to_bf16<<<tg, 256, 0, stream>>>(Wo, WoT, E, E);

  // 3) QKV projections (WMMA GEMM), scattered to attention-friendly layouts
  dim3 gg(E / 128, Mrows / 128);   // (8, 32)
  gemm_wmma_bf16<<<gg, 256, 0, stream>>>(Xbf, WqT, bq, Qb,  Mrows, E, E, 1);
  gemm_wmma_bf16<<<gg, 256, 0, stream>>>(Xbf, WkT, bk, Kbf, Mrows, E, E, 1);
  gemm_wmma_bf16<<<gg, 256, 0, stream>>>(Xbf, WvT, bv, Vtb, Mrows, E, E, 2);

  // 4) flash attention (WMMA QK^T + PV, online softmax)
  flash_attn_wmma<<<Bb * H * (SEQ / 64), 128, 0, stream>>>(Qb, Kbf, Vtb, mask, Ctx);

  // 5) output projection, fp32 result straight into d_out
  gemm_wmma_bf16<<<gg, 256, 0, stream>>>(Ctx, WoT, bo, d_out, Mrows, E, E, 3);
}
